// TransformerBlock_1571958030616
// MI455X (gfx1250) — matrix operand order repaired and checked by reading, not compiled, not run on hardware
//
#include <hip/hip_runtime.h>

// ---------- types ----------
typedef __bf16  bf16_t;
typedef __attribute__((ext_vector_type(16))) __bf16 v16bf;
typedef __attribute__((ext_vector_type(8)))  __bf16 v8bf;
typedef __attribute__((ext_vector_type(8)))  float  v8f;

__device__ __forceinline__ bf16_t f2bf(float f) {
    unsigned int u = __builtin_bit_cast(unsigned int, f);
    unsigned int r = (u + 0x7FFFu + ((u >> 16) & 1u)) >> 16;
    return __builtin_bit_cast(bf16_t, (unsigned short)r);
}

__device__ __forceinline__ v8f zero8() {
    v8f z;
#pragma unroll
    for (int i = 0; i < 8; ++i) z[i] = 0.0f;
    return z;
}

// A-fragment (16x32 bf16): lane l holds row M=l&15; K = e + 16*(l>>4) (16 contiguous).
// p must point at element (row, kbase + 16*(l>>4)): elems 0..7 at p, 8..15 at p+8.
__device__ __forceinline__ v16bf load_frag_a(const bf16_t* p) {
    v8bf lo = *reinterpret_cast<const v8bf*>(p);
    v8bf hi = *reinterpret_cast<const v8bf*>(p + 8);
    v16bf r;
#pragma unroll
    for (int i = 0; i < 8; ++i) { r[i] = lo[i]; r[8 + i] = hi[i]; }
    return r;
}

// B-fragment (32x16 bf16): lane l holds col N=l&15; K = e + 16*(l>>4) (16 contiguous).
__device__ __forceinline__ v16bf load_frag_b(const bf16_t* p) {
    v8bf lo = *reinterpret_cast<const v8bf*>(p);
    v8bf hi = *reinterpret_cast<const v8bf*>(p + 8);
    v16bf r;
#pragma unroll
    for (int i = 0; i < 8; ++i) { r[i] = lo[i]; r[8 + i] = hi[i]; }
    return r;
}

__device__ __forceinline__ v8f wmma_bf16(v16bf a, v16bf b, v8f c) {
    return __builtin_amdgcn_wmma_f32_16x16x32_bf16(false, a, false, b, (short)0, c, false, false);
}

// ---------- weight convert + transpose: fp32 [rows,cols] -> bf16 [cols,rows] ----------
__global__ void convert_transpose(const float* __restrict__ in, bf16_t* __restrict__ out,
                                  int rows, int cols) {
    size_t i = (size_t)blockIdx.x * blockDim.x + threadIdx.x;
    size_t total = (size_t)rows * cols;
    if (i >= total) return;
    int r = (int)(i / cols);
    int c = (int)(i % cols);
    out[(size_t)c * rows + r] = f2bf(in[i]);
}

// ---------- layernorm over D=1024, output bf16 ----------
__global__ void ln_kernel(const float* __restrict__ x, const float* __restrict__ g,
                          const float* __restrict__ bta, bf16_t* __restrict__ out) {
    __shared__ float ss[256], sq[256];
    int row = blockIdx.x, t = threadIdx.x;
    const float* xr = x + (size_t)row * 1024;
    float s = 0.f, q = 0.f;
#pragma unroll
    for (int i = t; i < 1024; i += 256) { float v = xr[i]; s += v; q += v * v; }
    ss[t] = s; sq[t] = q;
    __syncthreads();
    for (int off = 128; off > 0; off >>= 1) {
        if (t < off) { ss[t] += ss[t + off]; sq[t] += sq[t + off]; }
        __syncthreads();
    }
    float mu = ss[0] * (1.0f / 1024.0f);
    float var = sq[0] * (1.0f / 1024.0f) - mu * mu;
    float rstd = rsqrtf(var + 1e-5f);
    bf16_t* orow = out + (size_t)row * 1024;
#pragma unroll
    for (int i = t; i < 1024; i += 256)
        orow[i] = f2bf((xr[i] - mu) * rstd * g[i] + bta[i]);
}

// ---------- WMMA GEMM v2: 8 waves/block, block tile 128x64, LDS double-buffered B ----------
// C[M,N] = A[M,K](bf16) @ Bt[N,K](bf16)
// flags: 1=bias, 2=gelu(exact), 4=residual add, 8=store bf16 (else fp32)
__global__ __launch_bounds__(256) void gemm_wmma(
        const bf16_t* __restrict__ A, const bf16_t* __restrict__ Bt,
        float* __restrict__ Cf, bf16_t* __restrict__ Cb,
        const float* __restrict__ bias, const float* __restrict__ res,
        int M, int N, int K, int flags) {
    // double-buffered B tile: 64 cols x 32 K of bf16 = 4KB per buffer
    __shared__ __align__(16) bf16_t bt[2][64][32];

    int tid  = threadIdx.x;
    int lane = tid & 31, wv = tid >> 5;
    int lh = lane >> 4, ln15 = lane & 15;
    int n0   = blockIdx.x * 64;
    int row0 = blockIdx.y * 128 + wv * 16;

    v8f acc[4];
#pragma unroll
    for (int j = 0; j < 4; ++j) acc[j] = zero8();

    const bf16_t* arow = A + (size_t)(row0 + ln15) * K;

    // cooperative staging pattern: thread t covers row sn (of 64), 8 K elems at sk
    int sn = tid >> 2;            // 0..63
    int sk = (tid & 3) * 8;       // 0,8,16,24
    const bf16_t* gsrc = Bt + (size_t)(n0 + sn) * K + sk;

    // prologue: stage K-step 0 into buffer 0
    *reinterpret_cast<v8bf*>(&bt[0][sn][sk]) = *reinterpret_cast<const v8bf*>(gsrc);
    __syncthreads();

    int buf = 0;
    for (int k0 = 0; k0 < K; k0 += 32) {
        // per-wave A fragment (global, b128 x2)
        v16bf a = load_frag_a(arow + k0 + 16 * lh);

        // register-prefetch next B tile while computing
        bool more = (k0 + 32) < K;
        v8bf nb;
        if (more) {
            nb = *reinterpret_cast<const v8bf*>(gsrc + k0 + 32);
            __builtin_prefetch(arow + k0 + 32, 0, 0);
        }

        // 4 WMMAs from LDS-resident B fragments
#pragma unroll
        for (int j = 0; j < 4; ++j) {
            v16bf bfr = load_frag_b(&bt[buf][16 * j + ln15][16 * lh]);
            acc[j] = wmma_bf16(a, bfr, acc[j]);
        }

        if (more) {
            // write the *other* buffer (no reader this iteration), then one barrier
            *reinterpret_cast<v8bf*>(&bt[buf ^ 1][sn][sk]) = nb;
            __syncthreads();
            buf ^= 1;
        }
    }

    // epilogue
#pragma unroll
    for (int j = 0; j < 4; ++j) {
        int col = n0 + 16 * j + ln15;
        float bi = (flags & 1) ? bias[col] : 0.0f;
#pragma unroll
        for (int r = 0; r < 8; ++r) {
            int m = row0 + r + 8 * lh;
            float x = acc[j][r] + bi;
            if (flags & 2) x = 0.5f * x * (1.0f + erff(x * 0.70710678118654752f));
            size_t idx = (size_t)m * N + col;
            if (flags & 4) x += res[idx];
            if (flags & 8) Cb[idx] = f2bf(x);
            else           Cf[idx] = x;
        }
    }
}

// ---------- RoPE + repack: qf/kf/vf fp32 [B,S,H*64] -> qb/kb bf16 [B,H,S,64], vt bf16 [B,H,64,S]
__global__ void rope_pack(const float* __restrict__ qf, const float* __restrict__ kf,
                          const float* __restrict__ vf,
                          bf16_t* __restrict__ qb, bf16_t* __restrict__ kb,
                          bf16_t* __restrict__ vt) {
    unsigned i = blockIdx.x * blockDim.x + threadIdx.x; // B*S*H*32 pair-threads
    int pr = i & 31;
    int h  = (i >> 5) & 15;
    int s  = (i >> 9) & 2047;
    int b  = (int)(i >> 20);
    size_t src = ((size_t)(b * 2048 + s)) * 1024 + h * 64 + 2 * pr;
    float qe = qf[src], qo = qf[src + 1];
    float ke = kf[src], ko = kf[src + 1];
    float ve = vf[src], vo = vf[src + 1];
    // inv_freq = 10000^(-pr/32)
    float freq = (float)s * expf(-(float)pr * (9.210340371976184f / 32.0f));
    float c = cosf(freq), sn = sinf(freq);
    float qre = qe * c - qo * sn, qro = qo * c + qe * sn;
    float kre = ke * c - ko * sn, kro = ko * c + ke * sn;
    int bh = b * 16 + h;
    size_t qi = ((size_t)bh * 2048 + s) * 64 + 2 * pr;
    qb[qi] = f2bf(qre); qb[qi + 1] = f2bf(qro);
    kb[qi] = f2bf(kre); kb[qi + 1] = f2bf(kro);
    size_t vi = ((size_t)bh * 64 + 2 * pr) * 2048 + s;
    vt[vi] = f2bf(ve); vt[vi + 2048] = f2bf(vo);
}

// ---------- flash attention: one wave per 16-query tile per (b,h) ----------
// qb/kb: [B,H,S,64] bf16 ; vt: [B,H,64,S] bf16 ; attn out: [B*S, 1024] bf16
__global__ void flash_attn(const bf16_t* __restrict__ qb, const bf16_t* __restrict__ kb,
                           const bf16_t* __restrict__ vt, bf16_t* __restrict__ attn) {
    __shared__ __align__(16) bf16_t pf[16][32];
    int lane = threadIdx.x;
    int lh = lane >> 4, ln15 = lane & 15;
    int tile = blockIdx.x;          // B*H*(S/16) = 4096
    int q16 = tile & 127;
    int h   = (tile >> 7) & 15;
    int b   = tile >> 11;
    int bh  = b * 16 + h;
    int q0  = q16 * 16;

    const bf16_t* qbase = qb + ((size_t)bh * 2048 + q0 + ln15) * 64;
    v16bf a0 = load_frag_a(qbase + 16 * lh);
    v16bf a1 = load_frag_a(qbase + 32 + 16 * lh);

    v8f o[4];
#pragma unroll
    for (int j = 0; j < 4; ++j) o[j] = zero8();
    float m[8], l[8];
#pragma unroll
    for (int r = 0; r < 8; ++r) { m[r] = -1e30f; l[r] = 0.0f; }
    const float sc = 0.125f; // 1/sqrt(64)

    for (int key0 = 0; key0 < 2048; key0 += 32) {
        // scores: two 16x16 tiles covering 32 keys
        const bf16_t* kb0 = kb + ((size_t)bh * 2048 + key0 + ln15) * 64;
        const bf16_t* kb1 = kb0 + (size_t)16 * 64;
        v8f s0 = zero8(), s1 = zero8();
        s0 = wmma_bf16(a0, load_frag_b(kb0 + 16 * lh), s0);
        s0 = wmma_bf16(a1, load_frag_b(kb0 + 32 + 16 * lh), s0);
        s1 = wmma_bf16(a0, load_frag_b(kb1 + 16 * lh), s1);
        s1 = wmma_bf16(a1, load_frag_b(kb1 + 32 + 16 * lh), s1);

        float alpha[8];
#pragma unroll
        for (int r = 0; r < 8; ++r) {
            float v0 = s0[r] * sc, v1 = s1[r] * sc;
            float rm = fmaxf(v0, v1);
            rm = fmaxf(rm, __shfl_xor(rm, 1, 32));
            rm = fmaxf(rm, __shfl_xor(rm, 2, 32));
            rm = fmaxf(rm, __shfl_xor(rm, 4, 32));
            rm = fmaxf(rm, __shfl_xor(rm, 8, 32));
            float mn = fmaxf(m[r], rm);
            alpha[r] = __expf(m[r] - mn);
            float p0 = __expf(v0 - mn);
            float p1 = __expf(v1 - mn);
            float ps = p0 + p1;
            ps += __shfl_xor(ps, 1, 32);
            ps += __shfl_xor(ps, 2, 32);
            ps += __shfl_xor(ps, 4, 32);
            ps += __shfl_xor(ps, 8, 32);
            pf[r + 8 * lh][ln15]      = f2bf(p0);
            pf[r + 8 * lh][16 + ln15] = f2bf(p1);
            l[r] = l[r] * alpha[r] + ps;
            m[r] = mn;
        }
#pragma unroll
        for (int j = 0; j < 4; ++j)
#pragma unroll
            for (int r = 0; r < 8; ++r) o[j][r] *= alpha[r];

        __syncthreads();
        v16bf pa = load_frag_a(&pf[ln15][0] + 16 * lh);
#pragma unroll
        for (int j = 0; j < 4; ++j) {
            const bf16_t* vb = vt + ((size_t)bh * 64 + 16 * j + ln15) * 2048 + key0 + 16 * lh;
            o[j] = wmma_bf16(pa, load_frag_b(vb), o[j]);
        }
        __syncthreads();
    }

#pragma unroll
    for (int r = 0; r < 8; ++r) l[r] = 1.0f / l[r];
#pragma unroll
    for (int j = 0; j < 4; ++j)
#pragma unroll
        for (int r = 0; r < 8; ++r) {
            int grow = b * 2048 + q0 + r + 8 * lh;
            attn[(size_t)grow * 1024 + h * 64 + 16 * j + ln15] = f2bf(o[j][r] * l[r]);
        }
}

// ---------- launch ----------
extern "C" void kernel_launch(void* const* d_in, const int* in_sizes, int n_in,
                              void* d_out, int out_size, void* d_ws, size_t ws_size,
                              hipStream_t stream) {
    (void)in_sizes; (void)n_in; (void)out_size; (void)ws_size;
    const float* inputs = (const float*)d_in[0];
    const float* ln1_g  = (const float*)d_in[1];
    const float* ln1_b  = (const float*)d_in[2];
    const float* Wq     = (const float*)d_in[3];
    const float* Wk     = (const float*)d_in[4];
    const float* Wv     = (const float*)d_in[5];
    const float* Wo     = (const float*)d_in[6];
    const float* ln2_g  = (const float*)d_in[7];
    const float* ln2_b  = (const float*)d_in[8];
    const float* Wfc2   = (const float*)d_in[9];
    const float* bfc2   = (const float*)d_in[10];
    const float* Wfc3   = (const float*)d_in[11];
    const float* bfc3   = (const float*)d_in[12];
    float* out = (float*)d_out;

    const size_t BS = 4096, D = 1024, F = 4096;
    char* w = (char*)d_ws;
    auto alloc = [&](size_t bytes) -> void* {
        void* p = (void*)w;
        w += (bytes + 255) & ~(size_t)255;
        return p;
    };
    bf16_t* xb     = (bf16_t*)alloc(BS * D * 2);
    bf16_t* wq_t   = (bf16_t*)alloc(D * D * 2);
    bf16_t* wk_t   = (bf16_t*)alloc(D * D * 2);
    bf16_t* wv_t   = (bf16_t*)alloc(D * D * 2);
    bf16_t* wo_t   = (bf16_t*)alloc(D * D * 2);
    bf16_t* wfc2_t = (bf16_t*)alloc(D * F * 2);
    bf16_t* wfc3_t = (bf16_t*)alloc(F * D * 2);
    float*  qf     = (float*) alloc(BS * D * 4);
    float*  kf     = (float*) alloc(BS * D * 4);
    float*  vf     = (float*) alloc(BS * D * 4);
    bf16_t* qbb    = (bf16_t*)alloc(BS * D * 2);
    bf16_t* kbb    = (bf16_t*)alloc(BS * D * 2);
    bf16_t* vtb    = (bf16_t*)alloc(BS * D * 2);
    bf16_t* attn   = (bf16_t*)alloc(BS * D * 2);
    float*  mlp_in = (float*) alloc(BS * D * 4);
    bf16_t* hb     = (bf16_t*)alloc(BS * D * 2);
    bf16_t* hmid   = (bf16_t*)alloc(BS * F * 2);

    // weights -> bf16, transposed to [N,K]
    {
        int n1 = (int)(D * D);
        convert_transpose<<<(n1 + 255) / 256, 256, 0, stream>>>(Wq, wq_t, (int)D, (int)D);
        convert_transpose<<<(n1 + 255) / 256, 256, 0, stream>>>(Wk, wk_t, (int)D, (int)D);
        convert_transpose<<<(n1 + 255) / 256, 256, 0, stream>>>(Wv, wv_t, (int)D, (int)D);
        convert_transpose<<<(n1 + 255) / 256, 256, 0, stream>>>(Wo, wo_t, (int)D, (int)D);
        int n2 = (int)(D * F);
        convert_transpose<<<(n2 + 255) / 256, 256, 0, stream>>>(Wfc2, wfc2_t, (int)D, (int)F);
        convert_transpose<<<(n2 + 255) / 256, 256, 0, stream>>>(Wfc3, wfc3_t, (int)F, (int)D);
    }

    // ln1
    ln_kernel<<<(unsigned)BS, 256, 0, stream>>>(inputs, ln1_g, ln1_b, xb);

    // q,k,v projections (fp32 out): grid = (N/64, M/128), 256 threads
    dim3 gqkv((unsigned)(D / 64), (unsigned)(BS / 128));
    gemm_wmma<<<gqkv, 256, 0, stream>>>(xb, wq_t, qf, nullptr, nullptr, nullptr,
                                        (int)BS, (int)D, (int)D, 0);
    gemm_wmma<<<gqkv, 256, 0, stream>>>(xb, wk_t, kf, nullptr, nullptr, nullptr,
                                        (int)BS, (int)D, (int)D, 0);
    gemm_wmma<<<gqkv, 256, 0, stream>>>(xb, wv_t, vf, nullptr, nullptr, nullptr,
                                        (int)BS, (int)D, (int)D, 0);

    // rope + repack (2*2048*16*32 = 2097152 threads)
    rope_pack<<<2097152 / 256, 256, 0, stream>>>(qf, kf, vf, qbb, kbb, vtb);

    // flash attention
    flash_attn<<<4096, 32, 0, stream>>>(qbb, kbb, vtb, attn);

    // output projection + residual -> mlp_in (fp32)
    gemm_wmma<<<gqkv, 256, 0, stream>>>(attn, wo_t, mlp_in, nullptr, nullptr, inputs,
                                        (int)BS, (int)D, (int)D, 4);

    // ln2
    ln_kernel<<<(unsigned)BS, 256, 0, stream>>>(mlp_in, ln2_g, ln2_b, hb);

    // fc2 + bias + gelu -> bf16
    dim3 gfc2((unsigned)(F / 64), (unsigned)(BS / 128));
    gemm_wmma<<<gfc2, 256, 0, stream>>>(hb, wfc2_t, nullptr, hmid, bfc2, nullptr,
                                        (int)BS, (int)F, (int)D, 1 | 2 | 8);

    // fc3 + bias + residual -> fp32 final output
    dim3 gfc3((unsigned)(D / 64), (unsigned)(BS / 128));
    gemm_wmma<<<gfc3, 256, 0, stream>>>(hmid, wfc3_t, out, nullptr, bfc3, mlp_in,
                                        (int)BS, (int)D, (int)F, 1 | 4);
}
